// Actor_72791105732800
// MI455X (gfx1250) — matrix operand order repaired and checked by reading, not compiled, not run on hardware
//
#include <hip/hip_runtime.h>
#include <hip/hip_bf16.h>

// ---------------- CDNA5 vector types ----------------
typedef __attribute__((ext_vector_type(16))) _Float16 v16h;
typedef __attribute__((ext_vector_type(8)))  float    v8f;
typedef __attribute__((ext_vector_type(4)))  int      v4i;

#define HDIM 512
#define G4   2048      // 4*H
#define VOCAB 50257
#define TENC 64
#define LDEC 128

#if defined(__gfx1250__) && __has_builtin(__builtin_amdgcn_global_load_async_to_lds_b128) && __has_builtin(__builtin_amdgcn_s_wait_asynccnt)
#define USE_ASYNC_LDS 1
typedef __attribute__((address_space(1))) v4i g_v4i;
typedef __attribute__((address_space(3))) v4i l_v4i;
#endif

// ---------------- helpers ----------------
__device__ __forceinline__ float wave_sum(float v) {
    for (int o = 16; o > 0; o >>= 1) v += __shfl_xor(v, o, 32);
    return v;
}
__device__ __forceinline__ float sigm(float x) { return 1.0f / (1.0f + expf(-x)); }

// Threefry-2x32-20 (jax PRNG core)
__device__ __forceinline__ void threefry2x32(unsigned k0, unsigned k1,
                                             unsigned c0, unsigned c1,
                                             unsigned& o0, unsigned& o1) {
    const unsigned ks2 = 0x1BD11BDAu ^ k0 ^ k1;
    unsigned x0 = c0 + k0, x1 = c1 + k1;
    const int ra[4] = {13, 15, 26, 6}, rb[4] = {17, 29, 16, 24};
#define RL(v, r) (((v) << (r)) | ((v) >> (32 - (r))))
    for (int i = 0; i < 4; i++) { x0 += x1; x1 = RL(x1, ra[i]); x1 ^= x0; }
    x0 += k1; x1 += ks2 + 1u;
    for (int i = 0; i < 4; i++) { x0 += x1; x1 = RL(x1, rb[i]); x1 ^= x0; }
    x0 += ks2; x1 += k0 + 2u;
    for (int i = 0; i < 4; i++) { x0 += x1; x1 = RL(x1, ra[i]); x1 ^= x0; }
    x0 += k0; x1 += k1 + 3u;
    for (int i = 0; i < 4; i++) { x0 += x1; x1 = RL(x1, rb[i]); x1 ^= x0; }
    x0 += k1; x1 += ks2 + 4u;
    for (int i = 0; i < 4; i++) { x0 += x1; x1 = RL(x1, ra[i]); x1 ^= x0; }
    x0 += ks2; x1 += k0 + 5u;
#undef RL
    o0 = x0; o1 = x1;
}

// ---------------- init: zero state, set SOS token ----------------
__global__ void k_init(float* h, float* c, float* attn_sum, int* tok) {
    int j = threadIdx.x;
    h[j] = 0.f; c[j] = 0.f; attn_sum[j] = 0.f;
    if (j == 0) *tok = 1; // SOS
}

// ---------------- fp32 -> f16 pre-swizzle into WMMA A layout -----------------
// 16-bit A-matrix 16x32 lane map used here (contiguous k, same order as B):
//   lane<16 : M=lane,   K = {0..15}  (+32*kblk)
//   lane>=16: M=lane-16,K = {16..31} (+32*kblk)
// Storage: [tile][kblk][lane][16 halves] -> each lane's 32B load is contiguous.
// Logical K axis: cols 0..511 from Wa, 512..1023 from Wb (if KB==32).
__global__ void k_cvt(const float* __restrict__ Wa, const float* __restrict__ Wb,
                      _Float16* __restrict__ dst, int NTt, int KB, int validR) {
    int idx  = blockIdx.x * 256 + threadIdx.x;
    int slots = KB * 32;
    int tile = idx / slots;
    int rem  = idx % slots;
    int kblk = rem / 32;
    int lane = rem % 32;
    if (tile >= NTt) return;
    int row = tile * 16 + (lane & 15);
    _Float16* d = dst + ((size_t)(tile * KB + kblk) * 32 + lane) * 16;
    for (int i = 0; i < 16; i++) {
        int K;
        if (lane < 16) K = kblk * 32 + i;
        else           K = kblk * 32 + 16 + i;
        float v = 0.f;
        if (row < validR)
            v = (K < HDIM) ? Wa[(size_t)row * HDIM + K]
                           : Wb[(size_t)row * HDIM + (K - HDIM)];
        d[i] = (_Float16)v;
    }
}

// ---------------- encoder input gates: xg[t][j] = Wih[j]@emb[x[t]] + bih + bhh
__global__ void k_enc_xg(const float* __restrict__ Wih,
                         const float* __restrict__ emb_in,
                         const int* __restrict__ x,
                         const float* __restrict__ bih,
                         const float* __restrict__ bhh,
                         float* __restrict__ xg) {
    int wid  = blockIdx.x * 8 + (threadIdx.x >> 5);
    int lane = threadIdx.x & 31;
    int t = wid / G4, j = wid % G4;
    if (t >= TENC) return;
    const float* e = emb_in + (size_t)x[t] * HDIM;
    const float* w = Wih + (size_t)j * HDIM;
    float s = 0.f;
    for (int k = lane; k < HDIM; k += 32) s += w[k] * e[k];
    s = wave_sum(s);
    if (lane == 0) xg[(size_t)t * G4 + j] = s + bih[j] + bhh[j];
}

// ---------------- generic wave-per-row GEMV (used for cW2 prep) --------------
__global__ void k_gemv(const float* __restrict__ W, int ld, int off,
                       const float* __restrict__ x,
                       const float* __restrict__ add,
                       float* __restrict__ y, int R) {
    int row  = blockIdx.x * 8 + (threadIdx.x >> 5);
    int lane = threadIdx.x & 31;
    if (row >= R) return;
    const float* w = W + (size_t)row * ld + off;
    float s = 0.f;
    for (int k = lane; k < HDIM; k += 32) s += w[k] * x[k];
    s = wave_sum(s);
    if (lane == 0) y[row] = s + add[row];
}

// ---------------- copy -------------------------------------------------------
__global__ void k_copy(float* dst, const float* src, int n) {
    int i = blockIdx.x * 256 + threadIdx.x;
    if (i < n) dst[i] = src[i];
}

// ============ fused encoder step: gates = WMMA(Whh16, h) + xg_t; LSTM ========
// Single workgroup (32 waves). 128 row-tiles, 4 per wave, K=512 (16 kblks).
__global__ __launch_bounds__(1024) void k_enc_step(
        const _Float16* __restrict__ W16,    // encWhh, swizzled [128][16][32][16]
        const float* __restrict__ xg_t,      // precomputed input gates + biases
        float* __restrict__ h, float* __restrict__ c,
        float* __restrict__ attn_sum) {
    __shared__ __align__(32) _Float16 xls[HDIM];
    __shared__ float gls[G4];
    int tid = threadIdx.x;
    if (tid < HDIM) xls[tid] = (_Float16)h[tid];
    __syncthreads();
    int wid = tid >> 5, lane = tid & 31;
    for (int i = 0; i < 4; i++) {
        int tile = wid * 4 + i;
        const _Float16* A = W16 + (size_t)tile * (16 * HDIM);
        v8f acc = {};
#pragma unroll
        for (int kb = 0; kb < 16; kb++) {
            v16h a = *(const v16h*)(A + ((size_t)kb * 32 + lane) * 16);
            v16h b = {};
            if (lane == 0)  b = *(const v16h*)&xls[kb * 32];
            if (lane == 16) b = *(const v16h*)&xls[kb * 32 + 16];
            acc = __builtin_amdgcn_wmma_f32_16x16x32_f16(
                false, a, false, b, (short)0, acc, false, false);
        }
        if (lane == 0 || lane == 16) {
            int rb = tile * 16 + ((lane == 16) ? 8 : 0);
            for (int r = 0; r < 8; r++) gls[rb + r] = acc[r];
        }
    }
    __syncthreads();
    if (tid < HDIM) {
        float gi = sigm(gls[tid]            + xg_t[tid]);
        float gf = sigm(gls[HDIM + tid]     + xg_t[HDIM + tid]);
        float gg = tanhf(gls[2 * HDIM + tid] + xg_t[2 * HDIM + tid]);
        float go = sigm(gls[3 * HDIM + tid] + xg_t[3 * HDIM + tid]);
        float c2 = gf * c[tid] + gi * gg;
        float h2 = go * tanhf(c2);
        c[tid] = c2; h[tid] = h2;
        attn_sum[tid] += h2;
    }
}

// ============ fused decoder step: comb -> WMMA gates (K=1024) -> LSTM ========
// Single workgroup (32 waves). comb in fp32 VALU; gates via WMMA on
// decW16 = [dec_Wih | dec_Whh] (K=1024, 32 kblks), x = [comb ; h] in LDS.
__global__ __launch_bounds__(1024) void k_dec_step(
        const _Float16* __restrict__ W16,    // [128][32][32][16]
        const float* __restrict__ combW,     // [512, 1024]
        const float* __restrict__ dec_emb,   // [V, 512]
        const int* __restrict__ tok,
        const float* __restrict__ cW2,       // comb_W[:,512:]@attn_sum + comb_b
        const float* __restrict__ bih, const float* __restrict__ bhh,
        float* __restrict__ h, float* __restrict__ c,
        float* __restrict__ states_row,      // states[s+1]
        _Float16* __restrict__ h16) {        // for logits kernel
    __shared__ __align__(32) _Float16 xls[2 * HDIM];  // [comb ; h] as f16
    __shared__ float gls[G4];
    int tid = threadIdx.x;
    int wid = tid >> 5, lane = tid & 31;
    if (tid < HDIM) xls[HDIM + tid] = (_Float16)h[tid];
    // comb = relu(comb_W[:, :512] @ dec_emb[tok] + cW2)
    const float* e = dec_emb + (size_t)(*tok) * HDIM;
    for (int r = 0; r < 16; r++) {
        int row = wid * 16 + r;
        const float* w = combW + (size_t)row * (2 * HDIM);
        float s = 0.f;
        for (int k = lane; k < HDIM; k += 32) s += w[k] * e[k];
        s = wave_sum(s);
        if (lane == 0) xls[row] = (_Float16)fmaxf(s + cW2[row], 0.f);
    }
    __syncthreads();
    for (int i = 0; i < 4; i++) {
        int tile = wid * 4 + i;
        const _Float16* A = W16 + (size_t)tile * (16 * 2 * HDIM);
        v8f acc = {};
#pragma unroll
        for (int kb = 0; kb < 32; kb++) {
            v16h a = *(const v16h*)(A + ((size_t)kb * 32 + lane) * 16);
            v16h b = {};
            if (lane == 0)  b = *(const v16h*)&xls[kb * 32];
            if (lane == 16) b = *(const v16h*)&xls[kb * 32 + 16];
            acc = __builtin_amdgcn_wmma_f32_16x16x32_f16(
                false, a, false, b, (short)0, acc, false, false);
        }
        if (lane == 0 || lane == 16) {
            int rb = tile * 16 + ((lane == 16) ? 8 : 0);
            for (int r = 0; r < 8; r++) gls[rb + r] = acc[r] + bih[rb + r] + bhh[rb + r];
        }
    }
    __syncthreads();
    if (tid < HDIM) {
        float gi = sigm(gls[tid]);
        float gf = sigm(gls[HDIM + tid]);
        float gg = tanhf(gls[2 * HDIM + tid]);
        float go = sigm(gls[3 * HDIM + tid]);
        float c2 = gf * c[tid] + gi * gg;
        float h2 = go * tanhf(c2);
        c[tid] = c2; h[tid] = h2;
        states_row[tid] = h2;
        h16[tid] = (_Float16)h2;
    }
}

// ---------------- output projection GEMV via WMMA f16 ------------------------
// One wave per 16-row tile; weights streamed (L2-resident f16 copy).
// Async path: GLOBAL_LOAD_ASYNC_TO_LDS_B128 double-buffer + s_wait_asynccnt.
__global__ __launch_bounds__(256) void k_logits_wmma(
        const _Float16* __restrict__ outW16,
        const _Float16* __restrict__ h16,
        const float* __restrict__ out_b,
        float* __restrict__ logits, int NT) {
    int wave = threadIdx.x >> 5;
    int lane = threadIdx.x & 31;
    int tile = blockIdx.x * 8 + wave;
    if (tile >= NT) return;                        // wave-uniform exit
    const _Float16* A = outW16 + (size_t)tile * (16 * HDIM);
    v8f acc = {};
#ifdef USE_ASYNC_LDS
    __shared__ __align__(32) _Float16 stage[2][8][32][16];   // 16 KB
    auto issue = [&](int kb, int buf) {
        g_v4i* src = (g_v4i*)(A + ((size_t)kb * 32 + lane) * 16);
        l_v4i* dst = (l_v4i*)&stage[buf][wave][lane][0];
        __builtin_amdgcn_global_load_async_to_lds_b128(src,     dst,     0, 0);
        __builtin_amdgcn_global_load_async_to_lds_b128(src + 1, dst + 1, 0, 0);
    };
    issue(0, 0);
#pragma unroll
    for (int kb = 0; kb < 15; kb++) {
        issue(kb + 1, (kb + 1) & 1);
        __builtin_amdgcn_s_wait_asynccnt(2);       // chunk kb arrived (in-order)
        v16h a = *(const v16h*)&stage[kb & 1][wave][lane][0];
        v16h b = {};
        if (lane == 0)  b = *(const v16h*)(h16 + kb * 32);
        if (lane == 16) b = *(const v16h*)(h16 + kb * 32 + 16);
        acc = __builtin_amdgcn_wmma_f32_16x16x32_f16(
            false, a, false, b, (short)0, acc, false, false);
    }
    {
        __builtin_amdgcn_s_wait_asynccnt(0);
        v16h a = *(const v16h*)&stage[15 & 1][wave][lane][0];
        v16h b = {};
        if (lane == 0)  b = *(const v16h*)(h16 + 15 * 32);
        if (lane == 16) b = *(const v16h*)(h16 + 15 * 32 + 16);
        acc = __builtin_amdgcn_wmma_f32_16x16x32_f16(
            false, a, false, b, (short)0, acc, false, false);
    }
#else
#pragma unroll
    for (int kblk = 0; kblk < 16; kblk++) {
        v16h a = *(const v16h*)(A + ((size_t)kblk * 32 + lane) * 16);
        __builtin_prefetch(A + ((size_t)(kblk + 4) * 32 + lane) * 16, 0, 0);
        v16h b = {};
        if (lane == 0)  b = *(const v16h*)(h16 + kblk * 32);
        if (lane == 16) b = *(const v16h*)(h16 + kblk * 32 + 16);
        acc = __builtin_amdgcn_wmma_f32_16x16x32_f16(
            false, a, false, b, (short)0, acc, false, false);
    }
#endif
    if (lane == 0 || lane == 16) {
        int rbase = tile * 16 + ((lane == 16) ? 8 : 0);
        for (int r = 0; r < 8; r++) {
            int row = rbase + r;
            if (row < VOCAB) logits[row] = acc[r] + out_b[row];
        }
    }
}

// ---------------- softmax + gumbel-argmax categorical sample ----------------
__global__ void k_sample(const float* __restrict__ logits,
                         const int* __restrict__ mask,
                         float* __restrict__ prob_out,
                         float* __restrict__ act_out,
                         int* __restrict__ tok, int step) {
    __shared__ float sf[1024];
    __shared__ int   si[1024];
    __shared__ float ssum, smax;
    int tid = threadIdx.x;
    float m = -INFINITY;
    for (int v = tid; v < VOCAB; v += 1024) m = fmaxf(m, logits[v]);
    sf[tid] = m; __syncthreads();
    for (int s = 512; s > 0; s >>= 1) {
        if (tid < s) sf[tid] = fmaxf(sf[tid], sf[tid + s]);
        __syncthreads();
    }
    if (tid == 0) smax = sf[0];
    __syncthreads();
    float maxv = smax;
    unsigned sk0, sk1;
    threefry2x32(0u, 42u, (unsigned)(step + 1), 0x9E3779B9u, sk0, sk1);
    float lsum = 0.f, bestv = -INFINITY;
    int besti = VOCAB;
    for (int v = tid; v < VOCAB; v += 1024) {
        float lg = logits[v];
        lsum += expf(lg - maxv);
        if (mask[v] != 0) {
            unsigned r0, r1;
            threefry2x32(sk0, sk1, (unsigned)v, 0u, r0, r1);
            float u   = (float)(r0 >> 8) * (1.0f / 16777216.0f) + (1.0f / 33554432.0f);
            float gum = -logf(-logf(u));
            float val = lg + gum;
            if (val > bestv || (val == bestv && v < besti)) { bestv = val; besti = v; }
        }
    }
    sf[tid] = lsum; __syncthreads();
    for (int s = 512; s > 0; s >>= 1) {
        if (tid < s) sf[tid] += sf[tid + s];
        __syncthreads();
    }
    if (tid == 0) ssum = sf[0];
    __syncthreads();
    sf[tid] = bestv; si[tid] = besti; __syncthreads();
    for (int s = 512; s > 0; s >>= 1) {
        if (tid < s) {
            float a = sf[tid], b = sf[tid + s];
            int ia = si[tid], ib = si[tid + s];
            if (b > a || (b == a && ib < ia)) { sf[tid] = b; si[tid] = ib; }
        }
        __syncthreads();
    }
    if (tid == 0) {
        int a = si[0]; if (a >= VOCAB) a = 0;
        float p = expf(logits[a] - maxv) / ssum * (mask[a] ? 1.0f : 0.0f);
        prob_out[step] = p;
        act_out[step]  = (float)a;
        *tok = a;
    }
}

// ---------------- host orchestration ----------------------------------------
extern "C" void kernel_launch(void* const* d_in, const int* in_sizes, int n_in,
                              void* d_out, int out_size, void* d_ws, size_t ws_size,
                              hipStream_t stream) {
    const int*   x        = (const int*)  d_in[0];
    const int*   mask     = (const int*)  d_in[1];
    const float* emb_in   = (const float*)d_in[2];
    const float* enc_Wih  = (const float*)d_in[3];
    const float* enc_Whh  = (const float*)d_in[4];
    const float* enc_bih  = (const float*)d_in[5];
    const float* enc_bhh  = (const float*)d_in[6];
    const float* dec_emb  = (const float*)d_in[7];
    // d_in[8]=attn_W, d_in[9]=attn_b: unused — softmax over a size-1 axis makes
    // attention weights all-ones, so attn_applied == colsum(encoder_outputs).
    const float* comb_W   = (const float*)d_in[10];
    const float* comb_b   = (const float*)d_in[11];
    const float* dec_Wih  = (const float*)d_in[12];
    const float* dec_Whh  = (const float*)d_in[13];
    const float* dec_bih  = (const float*)d_in[14];
    const float* dec_bhh  = (const float*)d_in[15];
    const float* out_W    = (const float*)d_in[16];
    const float* out_b    = (const float*)d_in[17];

    float* states  = (float*)d_out;                        // [129, 512]
    float* actions = (float*)d_out + (LDEC + 1) * HDIM;    // [128]
    float* probs   = actions + LDEC;                       // [128]

    const int NT  = (VOCAB + 15) / 16;                     // 3142 vocab row tiles
    const int NTG = G4 / 16;                               // 128 gate row tiles

    char* base = (char*)d_ws;
    size_t off = 0;
    auto carve = [&](size_t bytes) -> void* {
        void* p = base + off;
        off = (off + bytes + 255) & ~(size_t)255;
        return p;
    };
    _Float16* outW16   = (_Float16*)carve((size_t)NT * 16 * HDIM * 2);   // 51.5 MB
    _Float16* encW16   = (_Float16*)carve((size_t)G4 * HDIM * 2);        // 2 MB
    _Float16* decW16   = (_Float16*)carve((size_t)G4 * 2 * HDIM * 2);    // 4 MB
    float*    xg       = (float*)carve((size_t)TENC * G4 * 4);
    float*    h        = (float*)carve(HDIM * 4);
    float*    c        = (float*)carve(HDIM * 4);
    _Float16* h16      = (_Float16*)carve(HDIM * 2);
    float*    attn_sum = (float*)carve(HDIM * 4);
    float*    cW2      = (float*)carve(HDIM * 4);
    float*    logits   = (float*)carve((size_t)VOCAB * 4);
    int*      tok      = (int*)carve(256);
    (void)ws_size; (void)n_in; (void)in_sizes; (void)out_size;

    // 0) init state
    k_init<<<1, HDIM, 0, stream>>>(h, c, attn_sum, tok);

    // 1) one-time f32->f16 pre-swizzles (all L2-resident afterwards)
    k_cvt<<<(NT * 16 * 32 + 255) / 256, 256, 0, stream>>>(out_W, nullptr, outW16,
                                                          NT, 16, VOCAB);
    k_cvt<<<(NTG * 16 * 32 + 255) / 256, 256, 0, stream>>>(enc_Whh, nullptr, encW16,
                                                           NTG, 16, G4);
    k_cvt<<<(NTG * 32 * 32 + 255) / 256, 256, 0, stream>>>(dec_Wih, dec_Whh, decW16,
                                                           NTG, 32, G4);

    // 2) encoder input-gate GEMM (batched over all 64 tokens)
    k_enc_xg<<<(TENC * G4) / 8, 256, 0, stream>>>(enc_Wih, emb_in, x,
                                                  enc_bih, enc_bhh, xg);

    // 3) encoder recurrence: one fused WMMA+LSTM node per step
    for (int t = 0; t < TENC; t++)
        k_enc_step<<<1, 1024, 0, stream>>>(encW16, xg + (size_t)t * G4,
                                           h, c, attn_sum);

    // 4) constant attention path: cW2 = comb_W[:,512:] @ attn_sum + comb_b
    k_gemv<<<HDIM / 8, 256, 0, stream>>>(comb_W, 2 * HDIM, HDIM,
                                         attn_sum, comb_b, cW2, HDIM);
    k_copy<<<(HDIM + 255) / 256, 256, 0, stream>>>(states, h, HDIM);  // states[0]

    // 5) decoder: 3 nodes per step (fused step, WMMA logits, sample)
    for (int s = 0; s < LDEC; s++) {
        k_dec_step<<<1, 1024, 0, stream>>>(decW16, comb_W, dec_emb, tok, cW2,
                                           dec_bih, dec_bhh, h, c,
                                           states + (size_t)(s + 1) * HDIM, h16);
        k_logits_wmma<<<(NT + 7) / 8, 256, 0, stream>>>(outW16, h16, out_b,
                                                        logits, NT);
        k_sample<<<1, 1024, 0, stream>>>(logits, mask, probs, actions, tok, s);
    }
}
